// GNNModel_50242527429113
// MI455X (gfx1250) — compile-verified, matrix operand order checked by reading, not run on hardware
//
#include <hip/hip_runtime.h>
#include <cstddef>

// Problem constants (fixed by the reference)
#define NB   128
#define SS   512
#define FIN  64
#define HH   4
#define CC   128
#define NCLS 4
#define NN   (NB * SS)   // 65536
#define HC   512

typedef __bf16 bf16_t;
typedef bf16_t        v16bf __attribute__((ext_vector_type(16)));
typedef float         v8f   __attribute__((ext_vector_type(8)));
typedef float         v4f   __attribute__((ext_vector_type(4)));
typedef unsigned int  v4u   __attribute__((ext_vector_type(4)));
typedef unsigned int  v2u   __attribute__((ext_vector_type(2)));

union FragU { v4u u[2]; v16bf v; };

__device__ __forceinline__ unsigned short f2bf(float f) {
  unsigned int u = __float_as_uint(f);
  u += 0x7FFFu + ((u >> 16) & 1u);   // round-to-nearest-even
  return (unsigned short)(u >> 16);
}

// ---------------------------------------------------------------------------
// f32 -> bf16 elementwise convert
// ---------------------------------------------------------------------------
__global__ void cvt_bf16_kernel(const float* __restrict__ in,
                                unsigned short* __restrict__ out, int n) {
  int i = blockIdx.x * blockDim.x + threadIdx.x;
  if (i < n) out[i] = f2bf(in[i]);
}

// f32 [K,Nc] row-major -> bf16 [Nc,K] row-major (transpose + convert)
__global__ void cvt_bf16_transpose_kernel(const float* __restrict__ in,
                                          unsigned short* __restrict__ out,
                                          int K, int Nc) {
  int i = blockIdx.x * blockDim.x + threadIdx.x;
  if (i >= K * Nc) return;
  int k = i / Nc, j = i - k * Nc;
  out[(size_t)j * K + k] = f2bf(in[i]);
}

// ---------------------------------------------------------------------------
// bf16 WMMA GEMM:  C[M,Nc] = A[M,K] @ Bt[Nc,K]^T   (f32 accum/out)
// wave = 16 rows x 128 cols (8 WMMA tiles); block = 8 waves = 128 rows
// A-fragment per-lane layout (16-bit A 16x32): lane l holds row m=l&15,
// k-runs [koff, koff+8) and [koff+16, koff+24) with koff = (l&16)?8:0 ->
// two contiguous 16B loads. B fragment = row n of Bt, same k pattern.
// ---------------------------------------------------------------------------
#define GEMM_NT 8
__global__ __launch_bounds__(256)
void wmma_gemm_bf16_kernel(const unsigned short* __restrict__ A,
                           const unsigned short* __restrict__ Bt,
                           float* __restrict__ C,
                           int M, int K, int Nc) {
  const int lane = threadIdx.x & 31;
  const int wave = threadIdx.x >> 5;
  const int row0 = (blockIdx.x * 8 + wave) * 16;
  const int colBase = blockIdx.y * (16 * GEMM_NT);
  if (row0 >= M) return;

  const int m    = lane & 15;
  const int koff = (lane & 16) ? 8 : 0;
  const int rOff = (lane & 16) ? 8 : 0;

  v8f acc[GEMM_NT];
#pragma unroll
  for (int j = 0; j < GEMM_NT; ++j) acc[j] = (v8f){0.f,0.f,0.f,0.f,0.f,0.f,0.f,0.f};

  const unsigned short* aRow = A + (size_t)(row0 + m) * K + koff;

  for (int k0 = 0; k0 < K; k0 += 32) {
    FragU a;
    a.u[0] = *(const v4u*)(aRow + k0);
    a.u[1] = *(const v4u*)(aRow + k0 + 16);
    if (k0 + 32 < K) __builtin_prefetch(aRow + k0 + 32, 0, 3);
#pragma unroll
    for (int j = 0; j < GEMM_NT; ++j) {
      const unsigned short* bRow =
          Bt + (size_t)(colBase + j * 16 + m) * K + koff + k0;
      FragU b;
      b.u[0] = *(const v4u*)(bRow);
      b.u[1] = *(const v4u*)(bRow + 16);
      acc[j] = __builtin_amdgcn_wmma_f32_16x16x32_bf16(
          /*neg_a=*/false, a.v, /*neg_b=*/false, b.v,
          /*c_mod=*/(short)0, acc[j], /*reuse_a=*/false, /*reuse_b=*/false);
    }
  }

  // C/D layout: VGPR r -> row r + rOff, col = lane&15
#pragma unroll
  for (int j = 0; j < GEMM_NT; ++j) {
    float* cPtr = C + (size_t)row0 * Nc + colBase + j * 16 + m;
#pragma unroll
    for (int r = 0; r < 8; ++r)
      cPtr[(size_t)(r + rOff) * Nc] = acc[j][r];
  }
}

// ---------------------------------------------------------------------------
// alpha_src[n,h] = <h[n,h,:], a_s[h,:]>, alpha_dst likewise. thread = (n,h).
// ---------------------------------------------------------------------------
__global__ void alpha_kernel(const float* __restrict__ h,
                             const float* __restrict__ a_s,
                             const float* __restrict__ a_d,
                             float* __restrict__ alpha_src,
                             float* __restrict__ alpha_dst) {
  int id = blockIdx.x * blockDim.x + threadIdx.x;
  if (id >= NN * HH) return;
  int n = id >> 2, hh = id & 3;
  const v4f* hp  = (const v4f*)(h + (size_t)n * HC + hh * CC);
  const v4f* asp = (const v4f*)(a_s + hh * CC);
  const v4f* adp = (const v4f*)(a_d + hh * CC);
  float ss = 0.f, sd = 0.f;
#pragma unroll 8
  for (int i = 0; i < CC / 4; ++i) {
    v4f hv = hp[i], av = asp[i], dv = adp[i];
    ss += hv.x * av.x + hv.y * av.y + hv.z * av.z + hv.w * av.w;
    sd += hv.x * dv.x + hv.y * dv.y + hv.z * dv.z + hv.w * dv.w;
  }
  alpha_src[id] = ss;
  alpha_dst[id] = sd;
}

// ---------------------------------------------------------------------------
// Chain-graph GAT attention + aggregation + bias + ReLU.
// Incoming edges of node n: {n-1 (if p>0), n+1 (if p<S-1), n}.
// block = 1 node, 128 threads (32 per head, 4 floats each).
// ---------------------------------------------------------------------------
template <int WRITE_BF16>
__global__ __launch_bounds__(128)
void gat_agg_kernel(const float* __restrict__ hpre,
                    const float* __restrict__ asrc,
                    const float* __restrict__ adst,
                    const float* __restrict__ bias,
                    float* __restrict__ out_f32,
                    unsigned short* __restrict__ out_bf16) {
  int n = blockIdx.x;
  int t = threadIdx.x;
  int hh = t >> 5;
  int p = n & (SS - 1);
  bool hasL = (p != 0), hasR = (p != SS - 1);

  float ad = adst[n * HH + hh];
  float aS = asrc[n * HH + hh] + ad;
  aS = aS > 0.f ? aS : 0.2f * aS;
  float mx = aS, aL = 0.f, aR = 0.f;
  if (hasL) { aL = asrc[(n - 1) * HH + hh] + ad; aL = aL > 0.f ? aL : 0.2f * aL; mx = fmaxf(mx, aL); }
  if (hasR) { aR = asrc[(n + 1) * HH + hh] + ad; aR = aR > 0.f ? aR : 0.2f * aR; mx = fmaxf(mx, aR); }
  float eS = __expf(aS - mx);
  float eL = hasL ? __expf(aL - mx) : 0.f;
  float eR = hasR ? __expf(aR - mx) : 0.f;
  float inv = 1.f / (eS + eL + eR);
  float wS = eS * inv, wL = eL * inv, wR = eR * inv;

  int hc = hh * CC + (t & 31) * 4;
  v4f acc = wS * (*(const v4f*)(hpre + (size_t)n * HC + hc));
  if (hasL) acc += wL * (*(const v4f*)(hpre + (size_t)(n - 1) * HC + hc));
  if (hasR) acc += wR * (*(const v4f*)(hpre + (size_t)(n + 1) * HC + hc));
  acc += *(const v4f*)(bias + hc);
  acc.x = fmaxf(acc.x, 0.f); acc.y = fmaxf(acc.y, 0.f);
  acc.z = fmaxf(acc.z, 0.f); acc.w = fmaxf(acc.w, 0.f);

  if (WRITE_BF16) {
    v2u pk;
    pk.x = (unsigned)f2bf(acc.x) | ((unsigned)f2bf(acc.y) << 16);
    pk.y = (unsigned)f2bf(acc.z) | ((unsigned)f2bf(acc.w) << 16);
    *(v2u*)(out_bf16 + (size_t)n * HC + hc) = pk;
  } else {
    *(v4f*)(out_f32 + (size_t)n * HC + hc) = acc;
  }
}

// ---------------------------------------------------------------------------
// scores[n] = <h2[n,:], w_pool> + b_pool   (one wave per node)
// ---------------------------------------------------------------------------
__global__ void pool_score_kernel(const float* __restrict__ h2,
                                  const float* __restrict__ w_pool,
                                  const float* __restrict__ b_pool,
                                  float* __restrict__ scores) {
  int gid = blockIdx.x * blockDim.x + threadIdx.x;
  int n = gid >> 5, lane = gid & 31;
  if (n >= NN) return;
  const float* row = h2 + (size_t)n * HC;
  float s = 0.f;
#pragma unroll
  for (int k = 0; k < HC / 32; ++k) {
    int hc = lane + 32 * k;
    s += row[hc] * w_pool[hc];
  }
#pragma unroll
  for (int off = 16; off > 0; off >>= 1) s += __shfl_xor(s, off, 32);
  if (lane == 0) scores[n] = s + b_pool[0];
}

// ---------------------------------------------------------------------------
// softmax over S per batch + weighted sum: pooled[b,hc] = sum_s w[s]*h2[b,s,hc]
// block = 1 batch (256 threads)
// ---------------------------------------------------------------------------
__global__ __launch_bounds__(256)
void pool_softmax_kernel(const float* __restrict__ h2,
                         const float* __restrict__ scores,
                         float* __restrict__ pooled) {
  __shared__ float w[SS];
  __shared__ float red[256];
  int b = blockIdx.x, t = threadIdx.x;
  float s0 = scores[b * SS + t];
  float s1 = scores[b * SS + t + 256];
  red[t] = fmaxf(s0, s1);
  __syncthreads();
  for (int off = 128; off > 0; off >>= 1) {
    if (t < off) red[t] = fmaxf(red[t], red[t + off]);
    __syncthreads();
  }
  float mx = red[0];
  __syncthreads();
  float e0 = __expf(s0 - mx), e1 = __expf(s1 - mx);
  red[t] = e0 + e1;
  __syncthreads();
  for (int off = 128; off > 0; off >>= 1) {
    if (t < off) red[t] += red[t + off];
    __syncthreads();
  }
  float inv = 1.f / red[0];
  w[t] = e0 * inv;
  w[t + 256] = e1 * inv;
  __syncthreads();

  const float* base = h2 + (size_t)b * SS * HC;
  float a0 = 0.f, a1 = 0.f;
  for (int s = 0; s < SS; ++s) {
    float ws = w[s];
    a0 += ws * base[(size_t)s * HC + t];
    a1 += ws * base[(size_t)s * HC + t + 256];
  }
  pooled[b * HC + t] = a0;
  pooled[b * HC + t + 256] = a1;
}

// ---------------------------------------------------------------------------
// out[b,c] = <pooled[b,:], W_fc[:,c]> + b_fc[c]   (one wave per output)
// ---------------------------------------------------------------------------
__global__ void fc_kernel(const float* __restrict__ pooled,
                          const float* __restrict__ Wfc,
                          const float* __restrict__ bfc,
                          float* __restrict__ out) {
  int gid = blockIdx.x * blockDim.x + threadIdx.x;
  int wv = gid >> 5, lane = gid & 31;
  if (wv >= NB * NCLS) return;
  int b = wv >> 2, c = wv & 3;
  float s = 0.f;
#pragma unroll
  for (int k = 0; k < HC / 32; ++k) {
    int hc = lane + 32 * k;
    s += pooled[b * HC + hc] * Wfc[hc * NCLS + c];
  }
#pragma unroll
  for (int off = 16; off > 0; off >>= 1) s += __shfl_xor(s, off, 32);
  if (lane == 0) out[b * NCLS + c] = s + bfc[c];
}

// ---------------------------------------------------------------------------
extern "C" void kernel_launch(void* const* d_in, const int* in_sizes, int n_in,
                              void* d_out, int out_size, void* d_ws, size_t ws_size,
                              hipStream_t stream) {
  (void)in_sizes; (void)n_in; (void)out_size; (void)ws_size;

  const float* x     = (const float*)d_in[0];
  const float* W0    = (const float*)d_in[1];
  const float* as0   = (const float*)d_in[2];
  const float* ad0   = (const float*)d_in[3];
  const float* b0    = (const float*)d_in[4];
  const float* W1    = (const float*)d_in[5];
  const float* as1   = (const float*)d_in[6];
  const float* ad1   = (const float*)d_in[7];
  const float* b1    = (const float*)d_in[8];
  const float* wpool = (const float*)d_in[9];
  const float* bpool = (const float*)d_in[10];
  const float* Wfc   = (const float*)d_in[11];
  const float* bfc   = (const float*)d_in[12];
  float* out = (float*)d_out;

  char* ws = (char*)d_ws;
  size_t off = 0;
  auto alloc = [&](size_t bytes) -> char* {
    char* p = ws + off;
    off += (bytes + 255) & ~(size_t)255;
    return p;
  };

  unsigned short* x_bf   = (unsigned short*)alloc((size_t)NN * FIN * 2);
  unsigned short* W0t_bf = (unsigned short*)alloc((size_t)FIN * HC * 2);
  unsigned short* W1t_bf = (unsigned short*)alloc((size_t)HC * HC * 2);
  float*          h_pre  = (float*)alloc((size_t)NN * HC * 4);
  float*          a_src  = (float*)alloc((size_t)NN * HH * 4);
  float*          a_dst  = (float*)alloc((size_t)NN * HH * 4);
  unsigned short* h1_bf  = (unsigned short*)alloc((size_t)NN * HC * 2);
  float*          h2_out = (float*)alloc((size_t)NN * HC * 4);
  float*          scores = (float*)alloc((size_t)NN * 4);
  float*          pooled = (float*)alloc((size_t)NB * HC * 4);

  // --- precision conversion / weight transposes ---
  cvt_bf16_kernel<<<(NN * FIN + 255) / 256, 256, 0, stream>>>(x, x_bf, NN * FIN);
  cvt_bf16_transpose_kernel<<<(FIN * HC + 255) / 256, 256, 0, stream>>>(W0, W0t_bf, FIN, HC);
  cvt_bf16_transpose_kernel<<<(HC * HC + 255) / 256, 256, 0, stream>>>(W1, W1t_bf, HC, HC);

  dim3 gemmGrid(NN / 128, HC / (16 * GEMM_NT));  // 512 x 4

  // --- layer 1: GEMM -> alphas -> aggregate (bf16 out) ---
  wmma_gemm_bf16_kernel<<<gemmGrid, 256, 0, stream>>>(x_bf, W0t_bf, h_pre, NN, FIN, HC);
  alpha_kernel<<<(NN * HH + 255) / 256, 256, 0, stream>>>(h_pre, as0, ad0, a_src, a_dst);
  gat_agg_kernel<1><<<NN, 128, 0, stream>>>(h_pre, a_src, a_dst, b0, nullptr, h1_bf);

  // --- layer 2: GEMM -> alphas -> aggregate (f32 out) ---
  wmma_gemm_bf16_kernel<<<gemmGrid, 256, 0, stream>>>(h1_bf, W1t_bf, h_pre, NN, HC, HC);
  alpha_kernel<<<(NN * HH + 255) / 256, 256, 0, stream>>>(h_pre, as1, ad1, a_src, a_dst);
  gat_agg_kernel<0><<<NN, 128, 0, stream>>>(h_pre, a_src, a_dst, b1, h2_out, nullptr);

  // --- attentive pooling + FC ---
  pool_score_kernel<<<(NN * 32 + 255) / 256, 256, 0, stream>>>(h2_out, wpool, bpool, scores);
  pool_softmax_kernel<<<NB, 256, 0, stream>>>(h2_out, scores, pooled);
  fc_kernel<<<(NB * NCLS * 32 + 255) / 256, 256, 0, stream>>>(pooled, Wfc, bfc, out);
}
